// GNN_prelu_32822140076345
// MI455X (gfx1250) — matrix operand order repaired and checked by reading, not compiled, not run on hardware
//
#include <hip/hip_runtime.h>

typedef _Float16 v16h __attribute__((ext_vector_type(16)));
typedef _Float16 v8h  __attribute__((ext_vector_type(8)));
typedef float    v8f  __attribute__((ext_vector_type(8)));

#define K_N_PFAS 50000
#define K_N_GW   300000
#define K_N_SW   100000
#define K_E_PG   1000000
#define K_E_GP   1000000
#define K_E_PS   500000
#define K_E_SP   500000

// ---------------- CDNA5 async global->LDS helpers ----------------
__device__ __forceinline__ unsigned gnn_lds_off(const void* p) {
    // LDS (addrspace 3) pointers are 32-bit offsets into the wave's LDS.
    return (unsigned)(uintptr_t)(__attribute__((address_space(3))) const void*)p;
}
__device__ __forceinline__ void gnn_async_ld_b128(unsigned lds, const float* g) {
    // GLOBAL_LOAD_ASYNC_TO_LDS_B128: VDST = LDS byte address, VADDR = 64-bit global.
    asm volatile("global_load_async_to_lds_b128 %0, %1, off"
                 :: "v"(lds), "v"((unsigned long long)(uintptr_t)g)
                 : "memory");
}
__device__ __forceinline__ void gnn_wait_async0() {
#if __has_builtin(__builtin_amdgcn_s_wait_asynccnt)
    __builtin_amdgcn_s_wait_asynccnt(0);
#else
    asm volatile("s_wait_asynccnt 0" ::: "memory");
#endif
}

// ---------------- edge in-degree counts (shared by both layers) ----------------
__global__ void gnn_count_kernel(const int* __restrict__ dst, float* __restrict__ cnt, int E) {
    int e = blockIdx.x * blockDim.x + threadIdx.x;
    if (e < E) atomicAdd(&cnt[dst[e]], 1.0f);
}

// ------------- scatter-add of D-wide feature rows (D = 1<<dshift) --------------
__global__ void gnn_scatter_add_kernel(const float* __restrict__ X, const int* __restrict__ src,
                                       const int* __restrict__ dst, float* __restrict__ agg,
                                       int E, int dshift) {
    int tid = blockIdx.x * blockDim.x + threadIdx.x;
    int cs  = dshift - 2;               // float4 chunks per row = 1<<cs
    int e   = tid >> cs;
    if (e >= E) return;
    int c = (tid & ((1 << cs) - 1)) << 2;
    int s = src[e], d = dst[e];
    const float4 v = *(const float4*)(X + ((size_t)s << dshift) + c);
    float* p = agg + ((size_t)d << dshift) + c;
    atomicAdd(p + 0, v.x); atomicAdd(p + 1, v.y);
    atomicAdd(p + 2, v.z); atomicAdd(p + 3, v.w);
}

// ---------------- mean = agg / max(cnt, 1) ----------------
__global__ void gnn_mean_div_kernel(float* __restrict__ agg, const float* __restrict__ cnt,
                                    int n, int dshift) {
    size_t idx = (size_t)blockIdx.x * blockDim.x + threadIdx.x;
    if (idx >= ((size_t)n << dshift)) return;
    int r = (int)(idx >> dshift);
    agg[idx] = agg[idx] / fmaxf(cnt[r], 1.0f);
}

// ---------------- WMMA GEMM: C[M,128] (+)= A[M,K] @ W[K,128] (+bias) ----------------
// 256 threads = 8 wave32; block tile 128(M) x 128(N); f16 inputs, f32 accumulate.
// A tiles are DMA'd global->LDS with GLOBAL_LOAD_ASYNC_TO_LDS_B128 (ASYNCcnt),
// double-buffered so the DMA of tile ks+1 overlaps the WMMAs of tile ks.
template <int K>
__global__ __launch_bounds__(256) void gnn_gemm_wmma(
    const float* __restrict__ A, const float* __restrict__ W,
    const float* __restrict__ bias, float* __restrict__ C,
    int M, int accum) {
    constexpr int NK = K / 32;
    __shared__ _Float16 sW[NK * 8 * 32 * 16];   // W pre-packed in B-fragment order (f16)
    __shared__ float    sA[2][128 * 32];        // double-buffered 128x32 f32 A tiles

    const int tid  = threadIdx.x;
    const int lane = tid & 31;
    const int wave = tid >> 5;
    const int Mb   = blockIdx.x * 128;

    // Pack W (f32 -> f16) into per-(kstep, ntile) WMMA B fragments.
    // B packed here in contiguous k order: lanes 0-15 hold K=0..15, lanes 16-31 hold K=16..31.
    for (int idx = tid; idx < NK * 8 * 32 * 16; idx += 256) {
        const int h  = idx & 15;
        const int ln = (idx >> 4) & 31;
        const int F  = idx >> 9;
        const int ks = F >> 3;
        const int nt = F & 7;
        const int k  = ks * 32 + (ln & 16) + h;
        const int n  = nt * 16 + (ln & 15);
        sW[idx] = (_Float16)W[k * 128 + n];
    }

    // Issue async DMA for one 128x32 tile (4 b128 chunks per thread).
    // Rows >= M are skipped: WMMA rows are independent and never stored.
    auto issue_tile = [&](int ks, int buf) {
#pragma unroll
        for (int j = 0; j < 4; ++j) {
            const int idx = tid + j * 256;   // float4 index in tile (8 per row)
            const int r   = idx >> 3;
            const int c4  = idx & 7;
            const int gr  = Mb + r;
            if (gr < M)
                gnn_async_ld_b128(gnn_lds_off(&sA[buf][r * 32 + c4 * 4]),
                                  A + (size_t)gr * K + ks * 32 + c4 * 4);
        }
    };
    issue_tile(0, 0);

    v8f acc[8] = {};

    for (int ks = 0; ks < NK; ++ks) {
        gnn_wait_async0();     // this wave's DMA for tile ks has landed in LDS
        __syncthreads();       // all waves' DMA landed; all waves done reading other buf
        if (ks + 1 < NK) issue_tile(ks + 1, (ks + 1) & 1);   // overlaps compute below

        // A fragment (16x32 16-bit): lane holds row (lane&15); halves 0..15 are
        // K = kb..kb+15, kb = (lane&16)?16:0.
        const int r  = lane & 15;
        const int kb = (lane & 16) ? 16 : 0;
        const float* row = &sA[ks & 1][(wave * 16 + r) * 32];
        const float4 f0 = *(const float4*)(row + kb);
        const float4 f1 = *(const float4*)(row + kb + 4);
        const float4 f2 = *(const float4*)(row + kb + 8);
        const float4 f3 = *(const float4*)(row + kb + 12);
        union { v16h v; _Float16 e[16]; } ua;
        ua.e[0]  = (_Float16)f0.x; ua.e[1]  = (_Float16)f0.y;
        ua.e[2]  = (_Float16)f0.z; ua.e[3]  = (_Float16)f0.w;
        ua.e[4]  = (_Float16)f1.x; ua.e[5]  = (_Float16)f1.y;
        ua.e[6]  = (_Float16)f1.z; ua.e[7]  = (_Float16)f1.w;
        ua.e[8]  = (_Float16)f2.x; ua.e[9]  = (_Float16)f2.y;
        ua.e[10] = (_Float16)f2.z; ua.e[11] = (_Float16)f2.w;
        ua.e[12] = (_Float16)f3.x; ua.e[13] = (_Float16)f3.y;
        ua.e[14] = (_Float16)f3.z; ua.e[15] = (_Float16)f3.w;
        const v16h a = ua.v;

#pragma unroll
        for (int nt = 0; nt < 8; ++nt) {
            const v16h b = *(const v16h*)&sW[((ks * 8 + nt) * 32 + lane) * 16];
            acc[nt] = __builtin_amdgcn_wmma_f32_16x16x32_f16(
                false, a, false, b, (short)0, acc[nt], false, false);
        }
    }

    // Epilogue: C/D layout — vgpr i, lane l: m = i + 8*(l>=16), n = l&15.
#pragma unroll
    for (int nt = 0; nt < 8; ++nt) {
        const int   n  = nt * 16 + (lane & 15);
        const float bv = bias ? bias[n] : 0.0f;
        const int   mh = (lane & 16) ? 8 : 0;
#pragma unroll
        for (int i = 0; i < 8; ++i) {
            const int m = Mb + wave * 16 + mh + i;
            if (m < M) {
                float v = acc[nt][i] + bv;
                if (accum) v += C[(size_t)m * 128 + n];
                C[(size_t)m * 128 + n] = v;
            }
        }
    }
}

// ---------------- BatchNorm batch stats (sum, sumsq per column) ----------------
__global__ void gnn_bn_stats_kernel(const float* __restrict__ X, float* __restrict__ stats, int n) {
    const int c = threadIdx.x;   // 128 threads, one per column
    float s = 0.f, q = 0.f;
    for (int r = blockIdx.x; r < n; r += gridDim.x) {
        const float v = X[(size_t)r * 128 + c];
        s += v; q += v * v;
    }
    atomicAdd(&stats[c], s);
    atomicAdd(&stats[128 + c], q);
}

__global__ void gnn_bn_relu_kernel(float* __restrict__ X, const float* __restrict__ stats,
                                   const float* __restrict__ g, const float* __restrict__ b, int n) {
    const size_t idx = (size_t)blockIdx.x * blockDim.x + threadIdx.x;
    if (idx >= (size_t)n * 128) return;
    const int   c     = (int)(idx & 127);
    const float inv_n = 1.0f / (float)n;
    const float m     = stats[c] * inv_n;
    const float var   = stats[128 + c] * inv_n - m * m;
    const float y     = (X[idx] - m) * (g[c] * rsqrtf(var + 1e-5f)) + b[c];
    X[idx] = fmaxf(y, 0.0f);
}

// ---------------- head: out = PReLU(G @ W[128,1] + b, a) ----------------
__global__ void gnn_head_prelu_kernel(const float* __restrict__ G, const float* __restrict__ Wv,
                                      const float* __restrict__ bptr, const float* __restrict__ aptr,
                                      float* __restrict__ out, int n) {
    const int r = blockIdx.x * blockDim.x + threadIdx.x;
    if (r >= n) return;
    const float4* gp = (const float4*)(G + (size_t)r * 128);
    const float4* wp = (const float4*)Wv;
    float s = 0.f;
#pragma unroll
    for (int j = 0; j < 32; ++j) {
        const float4 x = gp[j], w = wp[j];
        s = fmaf(x.x, w.x, s); s = fmaf(x.y, w.y, s);
        s = fmaf(x.z, w.z, s); s = fmaf(x.w, w.w, s);
    }
    s += bptr[0];
    const float a = aptr[0];
    out[r] = (s >= 0.f) ? s : a * s;
}

static inline unsigned gnn_cdiv(size_t a, size_t b) { return (unsigned)((a + b - 1) / b); }

extern "C" void kernel_launch(void* const* d_in, const int* in_sizes, int n_in,
                              void* d_out, int out_size, void* d_ws, size_t ws_size,
                              hipStream_t stream) {
    const float* x_pfas = (const float*)d_in[0];
    const float* x_gw   = (const float*)d_in[1];
    const float* x_sw   = (const float*)d_in[2];
    const int* src_pg = (const int*)d_in[3];
    const int* dst_pg = (const int*)d_in[4];
    const int* src_gp = (const int*)d_in[5];
    const int* dst_gp = (const int*)d_in[6];
    const int* src_ps = (const int*)d_in[7];
    const int* dst_ps = (const int*)d_in[8];
    const int* src_sp = (const int*)d_in[9];
    const int* dst_sp = (const int*)d_in[10];
    const float* Wl1 = (const float*)d_in[11];
    const float* bl1 = (const float*)d_in[12];
    const float* Wr1 = (const float*)d_in[13];
    const float* Wl2 = (const float*)d_in[14];
    const float* bl2 = (const float*)d_in[15];
    const float* Wr2 = (const float*)d_in[16];
    const float* bn_g_gw = (const float*)d_in[17];
    const float* bn_b_gw = (const float*)d_in[18];
    const float* bn_g_sw = (const float*)d_in[19];
    const float* bn_b_sw = (const float*)d_in[20];
    const float* W_gw = (const float*)d_in[21];
    const float* b_gw = (const float*)d_in[22];
    const float* W_sw = (const float*)d_in[23];
    const float* b_sw = (const float*)d_in[24];
    const float* a_gw = (const float*)d_in[25];
    const float* a_sw = (const float*)d_in[26];
    (void)in_sizes; (void)n_in; (void)out_size; (void)ws_size;

    float* ws = (float*)d_ws;
    size_t o = 0;
    float* cnt_gw    = ws + o; o += K_N_GW;
    float* cnt_pf_gp = ws + o; o += K_N_PFAS;
    float* cnt_pf_sp = ws + o; o += K_N_PFAS;
    float* cnt_sw    = ws + o; o += K_N_SW;
    float* stats     = ws + o; o += 256;
    float* agg       = ws + o; o += (size_t)(K_N_GW + 2 * K_N_PFAS + K_N_SW) * 128;
    float* h_gw      = ws + o; o += (size_t)K_N_GW * 128;
    float* h_pf      = ws + o; o += (size_t)K_N_PFAS * 128;
    float* h_sw      = ws + o; o += (size_t)K_N_SW * 128;
    float* g_gw      = ws + o; o += (size_t)K_N_GW * 128;
    float* g_sw      = ws + o; o += (size_t)K_N_SW * 128;

    float* out_gw = (float*)d_out;
    float* out_sw = out_gw + K_N_GW;
    float* g_pf   = out_gw + K_N_GW + K_N_SW;   // [N_PFAS,128], written directly

    // Layer-1 (64-wide) and layer-2 (128-wide) aggregation buffers share `agg`.
    float* a1_gw    = agg;
    float* a1_pf_gp = a1_gw    + (size_t)K_N_GW   * 64;
    float* a1_pf_sp = a1_pf_gp + (size_t)K_N_PFAS * 64;
    float* a1_sw    = a1_pf_sp + (size_t)K_N_PFAS * 64;
    float* a2_gw    = agg;
    float* a2_pf_gp = a2_gw    + (size_t)K_N_GW   * 128;
    float* a2_pf_sp = a2_pf_gp + (size_t)K_N_PFAS * 128;
    float* a2_sw    = a2_pf_sp + (size_t)K_N_PFAS * 128;

    // ---- in-degree counts (same edges both layers) ----
    hipMemsetAsync(cnt_gw, 0, ((size_t)(K_N_GW + 2 * K_N_PFAS + K_N_SW) + 256) * sizeof(float), stream);
    gnn_count_kernel<<<gnn_cdiv(K_E_PG, 256), 256, 0, stream>>>(dst_pg, cnt_gw, K_E_PG);
    gnn_count_kernel<<<gnn_cdiv(K_E_GP, 256), 256, 0, stream>>>(dst_gp, cnt_pf_gp, K_E_GP);
    gnn_count_kernel<<<gnn_cdiv(K_E_SP, 256), 256, 0, stream>>>(dst_sp, cnt_pf_sp, K_E_SP);
    gnn_count_kernel<<<gnn_cdiv(K_E_PS, 256), 256, 0, stream>>>(dst_ps, cnt_sw, K_E_PS);

    // ---- layer 1: aggregate (D=64) ----
    hipMemsetAsync(agg, 0, (size_t)(K_N_GW + 2 * K_N_PFAS + K_N_SW) * 64 * sizeof(float), stream);
    gnn_scatter_add_kernel<<<gnn_cdiv((size_t)K_E_PG * 16, 256), 256, 0, stream>>>(x_pfas, src_pg, dst_pg, a1_gw, K_E_PG, 6);
    gnn_scatter_add_kernel<<<gnn_cdiv((size_t)K_E_GP * 16, 256), 256, 0, stream>>>(x_gw,   src_gp, dst_gp, a1_pf_gp, K_E_GP, 6);
    gnn_scatter_add_kernel<<<gnn_cdiv((size_t)K_E_SP * 16, 256), 256, 0, stream>>>(x_sw,   src_sp, dst_sp, a1_pf_sp, K_E_SP, 6);
    gnn_scatter_add_kernel<<<gnn_cdiv((size_t)K_E_PS * 16, 256), 256, 0, stream>>>(x_pfas, src_ps, dst_ps, a1_sw, K_E_PS, 6);
    gnn_mean_div_kernel<<<gnn_cdiv((size_t)K_N_GW * 64, 256), 256, 0, stream>>>(a1_gw, cnt_gw, K_N_GW, 6);
    gnn_mean_div_kernel<<<gnn_cdiv((size_t)K_N_PFAS * 64, 256), 256, 0, stream>>>(a1_pf_gp, cnt_pf_gp, K_N_PFAS, 6);
    gnn_mean_div_kernel<<<gnn_cdiv((size_t)K_N_PFAS * 64, 256), 256, 0, stream>>>(a1_pf_sp, cnt_pf_sp, K_N_PFAS, 6);
    gnn_mean_div_kernel<<<gnn_cdiv((size_t)K_N_SW * 64, 256), 256, 0, stream>>>(a1_sw, cnt_sw, K_N_SW, 6);

    // ---- layer 1 GEMMs (K=64): h = mean@Wl + bl + x_dst@Wr, summed over relations ----
    gnn_gemm_wmma<64><<<gnn_cdiv(K_N_GW, 128), 256, 0, stream>>>(a1_gw, Wl1 + 0 * 8192, bl1 + 0 * 128, h_gw, K_N_GW, 0);
    gnn_gemm_wmma<64><<<gnn_cdiv(K_N_GW, 128), 256, 0, stream>>>(x_gw,  Wr1 + 0 * 8192, nullptr,       h_gw, K_N_GW, 1);
    gnn_gemm_wmma<64><<<gnn_cdiv(K_N_PFAS, 128), 256, 0, stream>>>(a1_pf_gp, Wl1 + 1 * 8192, bl1 + 1 * 128, h_pf, K_N_PFAS, 0);
    gnn_gemm_wmma<64><<<gnn_cdiv(K_N_PFAS, 128), 256, 0, stream>>>(x_pfas,   Wr1 + 1 * 8192, nullptr,       h_pf, K_N_PFAS, 1);
    gnn_gemm_wmma<64><<<gnn_cdiv(K_N_PFAS, 128), 256, 0, stream>>>(a1_pf_sp, Wl1 + 3 * 8192, bl1 + 3 * 128, h_pf, K_N_PFAS, 1);
    gnn_gemm_wmma<64><<<gnn_cdiv(K_N_PFAS, 128), 256, 0, stream>>>(x_pfas,   Wr1 + 3 * 8192, nullptr,       h_pf, K_N_PFAS, 1);
    gnn_gemm_wmma<64><<<gnn_cdiv(K_N_SW, 128), 256, 0, stream>>>(a1_sw, Wl1 + 2 * 8192, bl1 + 2 * 128, h_sw, K_N_SW, 0);
    gnn_gemm_wmma<64><<<gnn_cdiv(K_N_SW, 128), 256, 0, stream>>>(x_sw,  Wr1 + 2 * 8192, nullptr,       h_sw, K_N_SW, 1);

    // ---- BN + ReLU on h_gw / h_sw ----
    hipMemsetAsync(stats, 0, 256 * sizeof(float), stream);
    gnn_bn_stats_kernel<<<512, 128, 0, stream>>>(h_gw, stats, K_N_GW);
    gnn_bn_relu_kernel<<<gnn_cdiv((size_t)K_N_GW * 128, 256), 256, 0, stream>>>(h_gw, stats, bn_g_gw, bn_b_gw, K_N_GW);
    hipMemsetAsync(stats, 0, 256 * sizeof(float), stream);
    gnn_bn_stats_kernel<<<512, 128, 0, stream>>>(h_sw, stats, K_N_SW);
    gnn_bn_relu_kernel<<<gnn_cdiv((size_t)K_N_SW * 128, 256), 256, 0, stream>>>(h_sw, stats, bn_g_sw, bn_b_sw, K_N_SW);

    // ---- layer 2: aggregate (D=128) ----
    hipMemsetAsync(agg, 0, (size_t)(K_N_GW + 2 * K_N_PFAS + K_N_SW) * 128 * sizeof(float), stream);
    gnn_scatter_add_kernel<<<gnn_cdiv((size_t)K_E_PG * 32, 256), 256, 0, stream>>>(h_pf, src_pg, dst_pg, a2_gw, K_E_PG, 7);
    gnn_scatter_add_kernel<<<gnn_cdiv((size_t)K_E_GP * 32, 256), 256, 0, stream>>>(h_gw, src_gp, dst_gp, a2_pf_gp, K_E_GP, 7);
    gnn_scatter_add_kernel<<<gnn_cdiv((size_t)K_E_SP * 32, 256), 256, 0, stream>>>(h_sw, src_sp, dst_sp, a2_pf_sp, K_E_SP, 7);
    gnn_scatter_add_kernel<<<gnn_cdiv((size_t)K_E_PS * 32, 256), 256, 0, stream>>>(h_pf, src_ps, dst_ps, a2_sw, K_E_PS, 7);
    gnn_mean_div_kernel<<<gnn_cdiv((size_t)K_N_GW * 128, 256), 256, 0, stream>>>(a2_gw, cnt_gw, K_N_GW, 7);
    gnn_mean_div_kernel<<<gnn_cdiv((size_t)K_N_PFAS * 128, 256), 256, 0, stream>>>(a2_pf_gp, cnt_pf_gp, K_N_PFAS, 7);
    gnn_mean_div_kernel<<<gnn_cdiv((size_t)K_N_PFAS * 128, 256), 256, 0, stream>>>(a2_pf_sp, cnt_pf_sp, K_N_PFAS, 7);
    gnn_mean_div_kernel<<<gnn_cdiv((size_t)K_N_SW * 128, 256), 256, 0, stream>>>(a2_sw, cnt_sw, K_N_SW, 7);

    // ---- layer 2 GEMMs (K=128); g_pf goes straight into d_out ----
    gnn_gemm_wmma<128><<<gnn_cdiv(K_N_GW, 128), 256, 0, stream>>>(a2_gw, Wl2 + 0 * 16384, bl2 + 0 * 128, g_gw, K_N_GW, 0);
    gnn_gemm_wmma<128><<<gnn_cdiv(K_N_GW, 128), 256, 0, stream>>>(h_gw,  Wr2 + 0 * 16384, nullptr,       g_gw, K_N_GW, 1);
    gnn_gemm_wmma<128><<<gnn_cdiv(K_N_PFAS, 128), 256, 0, stream>>>(a2_pf_gp, Wl2 + 1 * 16384, bl2 + 1 * 128, g_pf, K_N_PFAS, 0);
    gnn_gemm_wmma<128><<<gnn_cdiv(K_N_PFAS, 128), 256, 0, stream>>>(h_pf,     Wr2 + 1 * 16384, nullptr,       g_pf, K_N_PFAS, 1);
    gnn_gemm_wmma<128><<<gnn_cdiv(K_N_PFAS, 128), 256, 0, stream>>>(a2_pf_sp, Wl2 + 3 * 16384, bl2 + 3 * 128, g_pf, K_N_PFAS, 1);
    gnn_gemm_wmma<128><<<gnn_cdiv(K_N_PFAS, 128), 256, 0, stream>>>(h_pf,     Wr2 + 3 * 16384, nullptr,       g_pf, K_N_PFAS, 1);
    gnn_gemm_wmma<128><<<gnn_cdiv(K_N_SW, 128), 256, 0, stream>>>(a2_sw, Wl2 + 2 * 16384, bl2 + 2 * 128, g_sw, K_N_SW, 0);
    gnn_gemm_wmma<128><<<gnn_cdiv(K_N_SW, 128), 256, 0, stream>>>(h_sw,  Wr2 + 2 * 16384, nullptr,       g_sw, K_N_SW, 1);

    // ---- BN + ReLU on g_gw / g_sw (g_pf is NOT normalized, matches reference) ----
    hipMemsetAsync(stats, 0, 256 * sizeof(float), stream);
    gnn_bn_stats_kernel<<<512, 128, 0, stream>>>(g_gw, stats, K_N_GW);
    gnn_bn_relu_kernel<<<gnn_cdiv((size_t)K_N_GW * 128, 256), 256, 0, stream>>>(g_gw, stats, bn_g_gw, bn_b_gw, K_N_GW);
    hipMemsetAsync(stats, 0, 256 * sizeof(float), stream);
    gnn_bn_stats_kernel<<<512, 128, 0, stream>>>(g_sw, stats, K_N_SW);
    gnn_bn_relu_kernel<<<gnn_cdiv((size_t)K_N_SW * 128, 256), 256, 0, stream>>>(g_sw, stats, bn_g_sw, bn_b_sw, K_N_SW);

    // ---- heads: linear [128->1] + PReLU ----
    gnn_head_prelu_kernel<<<gnn_cdiv(K_N_GW, 256), 256, 0, stream>>>(g_gw, W_gw, b_gw, a_gw, out_gw, K_N_GW);
    gnn_head_prelu_kernel<<<gnn_cdiv(K_N_SW, 256), 256, 0, stream>>>(g_sw, W_sw, b_sw, a_sw, out_sw, K_N_SW);
}
